// FEMREncoderLayer_20779051778399
// MI455X (gfx1250) — matrix operand order repaired and checked by reading, not compiled, not run on hardware
//
#include <hip/hip_runtime.h>
#include <hip/hip_bf16.h>
#include <stdint.h>

// ---------------------------------------------------------------------------
// FEMR encoder layer for MI455X (gfx1250), bf16 WMMA + TDM async-tensor path.
// T=2048, H=2048, NH=16, HD=128, I=8192
// ---------------------------------------------------------------------------

typedef __bf16 bf16;
typedef __attribute__((ext_vector_type(16))) __bf16 v16bf;
typedef __attribute__((ext_vector_type(8)))  __bf16 v8bf;
typedef __attribute__((ext_vector_type(8)))  float  v8f;
typedef __attribute__((ext_vector_type(4)))  unsigned int uint32x4;
typedef __attribute__((ext_vector_type(8)))  int  int32x8;
typedef __attribute__((ext_vector_type(4)))  int  int32x4;

#define T_DIM 2048
#define H_DIM 2048
#define NH    16
#define HD    128
#define I_DIM 8192
#define NT_COLS (3*H_DIM + 2*I_DIM)   // 22528
#define COMB_COLS (H_DIM + I_DIM)     // 10240

#if __has_builtin(__builtin_amdgcn_tensor_load_to_lds) && \
    __has_builtin(__builtin_amdgcn_s_wait_tensorcnt)
#define USE_TDM 1
#endif

// ---- workspace layout (bytes) ----
#define XN_OFF    ((size_t)0)
#define XN_BYTES  ((size_t)T_DIM * H_DIM * 2)                       // 8 MB
#define WINT_OFF  (XN_OFF + XN_BYTES)
#define WINT_BYTES ((size_t)NT_COLS * H_DIM * 2)                    // 92 MB
#define WOUTT_OFF (WINT_OFF + WINT_BYTES)
#define WOUTT_BYTES ((size_t)H_DIM * COMB_COLS * 2)                 // 42 MB
#define TBUF_OFF  (WOUTT_OFF + WOUTT_BYTES)
#define TBUF_BYTES ((size_t)T_DIM * NT_COLS * 4)                    // 184 MB
#define QKV_OFF   (TBUF_OFF + TBUF_BYTES)
#define QKV_ONE   ((size_t)NH * T_DIM * HD * 2)                     // 8.4 MB each
#define COMB_OFF  (QKV_OFF + 3*QKV_ONE)
#define COMB_BYTES ((size_t)T_DIM * COMB_COLS * 2)                  // 42 MB

// ---------------------------------------------------------------------------
// helpers
// ---------------------------------------------------------------------------
__device__ __forceinline__ v16bf make_frag(const bf16* p_lo, const bf16* p_hi) {
    v8bf lo = *(const v8bf*)p_lo;
    v8bf hi = *(const v8bf*)p_hi;
    return __builtin_shufflevector(lo, hi, 0,1,2,3,4,5,6,7,8,9,10,11,12,13,14,15);
}

__device__ __forceinline__ v8f wmma_bf16(v16bf a, v16bf b, v8f c) {
    return __builtin_amdgcn_wmma_f32_16x16x32_bf16(
        /*neg_a=*/false, a, /*neg_b=*/false, b,
        /*c_mod=*/(short)0, c, /*reuse_a=*/false, /*reuse_b=*/false);
}

#ifdef USE_TDM
// Issue one TDM 2D tile load (bf16 elements): tile 32(K) x 32(rows),
// LDS rows padded 64B data + 16B pad -> 80B stride (matches [.][40] tiles).
// All descriptor words forced wave-uniform via readfirstlane.
// NOTE: this toolchain ships the 6-arg builtin form:
//   (uint32x4 g0, int32x8 g1, int32x4 g2, int32x4 g3, int32x8 g4, i32 cpol)
__device__ __forceinline__ void tdm_load_tile32x32(
    const bf16* gptr, unsigned int lds_off, int K)
{
    const unsigned long long ga = (unsigned long long)(size_t)gptr;
    const unsigned int ga_lo = (unsigned int)__builtin_amdgcn_readfirstlane((int)(ga & 0xffffffffull));
    const unsigned int ga_hi = (unsigned int)__builtin_amdgcn_readfirstlane((int)(ga >> 32));
    const unsigned int lds   = (unsigned int)__builtin_amdgcn_readfirstlane((int)lds_off);

    uint32x4 g0;
    g0[0] = 1u;                                        // count=1, user mode
    g0[1] = lds;                                       // lds_addr (bytes)
    g0[2] = ga_lo;                                     // global_addr[31:0]
    g0[3] = (ga_hi & 0x01ffffffu) | (2u << 30);        // global_addr[56:32], type=2

    const unsigned int td0 = (unsigned int)K;          // tensor_dim0 (elems)
    const unsigned int td1 = 32u;                      // tensor_dim1
    const unsigned long long st0 = (unsigned long long)(unsigned int)K; // dim0 stride

    int32x8 g1;
    g1[0] = (int)((1u << 16)        // data_size = 2 bytes
                | (1u << 20)        // pad_enable
                | (3u << 22)        // pad_interval: 16 DWORDs (64B of data)
                | (3u << 25));      // pad_amount: 4 DWORDs (16B pad)
    g1[1] = (int)((td0 & 0xffffu) << 16);              // tensor_dim0[15:0]
    g1[2] = (int)((td0 >> 16) | ((td1 & 0xffffu) << 16)); // td0[31:16] | td1[15:0]
    g1[3] = (int)((td1 >> 16) | (32u << 16));          // td1[31:16] | tile_dim0=32
    g1[4] = (int)32u;                                  // tile_dim1=32, tile_dim2=0
    g1[5] = (int)(st0 & 0xffffffffull);                // tensor_dim0_stride[31:0]
    g1[6] = (int)((st0 >> 32) & 0xffffull);            // stride[47:32] | dim1_stride lo = 0
    g1[7] = 0;

    const int32x4 z4 = {0, 0, 0, 0};
    const int32x8 z8 = {0, 0, 0, 0, 0, 0, 0, 0};
    __builtin_amdgcn_tensor_load_to_lds(g0, g1, z4, z4, z8, 0);
}
#endif

// ---------------------------------------------------------------------------
// 1) RMSNorm + age feature override -> xn (bf16)
// ---------------------------------------------------------------------------
__global__ __launch_bounds__(256) void rmsnorm_kernel(
    const float* __restrict__ x, const float* __restrict__ ages,
    const float* __restrict__ w, bf16* __restrict__ xn)
{
    __shared__ float red[256];
    const int t = blockIdx.x, tid = threadIdx.x;
    const float* xr = x + (size_t)t * H_DIM;
    float s = 0.f;
    for (int c = tid; c < H_DIM; c += 256) { float v = xr[c]; s += v * v; }
    red[tid] = s; __syncthreads();
    for (int d = 128; d > 0; d >>= 1) { if (tid < d) red[tid] += red[tid + d]; __syncthreads(); }
    const float inv = rsqrtf(red[0] * (1.0f / H_DIM) + 1e-6f);
    const float age = ages[t];
    for (int c = tid; c < H_DIM; c += 256) {
        float v = xr[c] * inv * w[c];
        if (c == H_DIM - 2) v = age;
        if (c == H_DIM - 1) v = age * age;
        xn[(size_t)t * H_DIM + c] = (bf16)v;
    }
}

// ---------------------------------------------------------------------------
// 2) transpose + f32->bf16 convert:  in[R][C] -> out[C][R]
// ---------------------------------------------------------------------------
__global__ __launch_bounds__(256) void transpose_conv_kernel(
    const float* __restrict__ in, bf16* __restrict__ out, int R, int C)
{
    __shared__ float tile[32][33];
    const int tx = threadIdx.x, ty = threadIdx.y;
    const int bx = blockIdx.x, by = blockIdx.y;
    #pragma unroll
    for (int j = 0; j < 32; j += 8) {
        int r = by * 32 + ty + j, c = bx * 32 + tx;
        tile[ty + j][tx] = in[(size_t)r * C + c];
    }
    __syncthreads();
    #pragma unroll
    for (int j = 0; j < 32; j += 8) {
        int orow = bx * 32 + ty + j;  // original col
        int ocol = by * 32 + tx;      // original row
        out[(size_t)orow * R + ocol] = (bf16)tile[tx][ty + j];
    }
}

// ---------------------------------------------------------------------------
// 3) Tiled WMMA GEMM:  C[M][N] = A[M][K] (bf16) * BT[N][K]^T (bf16) + bias[N]
//    128x128 tile per 256-thread block (8 waves, 2x4 16x16 frags per wave).
//    TDM path: double-buffered LDS, each wave DMAs one 32x32 subtile per step,
//    overlapped with WMMA on the other buffer, synced with TENSORcnt.
// ---------------------------------------------------------------------------
__global__ __launch_bounds__(256) void gemm_bf16_kernel(
    const bf16* __restrict__ A, const bf16* __restrict__ BT,
    const float* __restrict__ bias, float* __restrict__ C,
    int M, int N, int K)
{
    const int tid  = threadIdx.x;
    const int lane = tid & 31, wid = tid >> 5;
    const int half = lane >> 4, l16 = lane & 15;
    const int wm = wid & 3;        // 4 waves along M (32 rows each)
    const int wn = wid >> 2;       // 2 waves along N (64 cols each)
    const int row0 = blockIdx.y * 128;
    const int col0 = blockIdx.x * 128;

    v8f acc[2][4];
    #pragma unroll
    for (int mi = 0; mi < 2; ++mi)
        #pragma unroll
        for (int ni = 0; ni < 4; ++ni) acc[mi][ni] = (v8f){0,0,0,0,0,0,0,0};

#ifdef USE_TDM
    __shared__ __attribute__((aligned(16))) bf16 sA[2][128][40];
    __shared__ __attribute__((aligned(16))) bf16 sB[2][128][40];

    // per-wave TDM subtile: waves 0-3 -> A rows, waves 4-7 -> B rows
    const int sub = wid & 3;
    const bf16* gbase = (wid < 4)
        ? (A  + (size_t)(row0 + sub * 32) * K)
        : (BT + (size_t)(col0 + sub * 32) * K);
    unsigned int lds_base[2];
    lds_base[0] = (unsigned int)(size_t)((wid < 4) ? &sA[0][sub * 32][0] : &sB[0][sub * 32][0]);
    lds_base[1] = (unsigned int)(size_t)((wid < 4) ? &sA[1][sub * 32][0] : &sB[1][sub * 32][0]);

    tdm_load_tile32x32(gbase, lds_base[0], K);   // prologue: tile k0=0 -> buf 0

    int buf = 0;
    for (int k0 = 0; k0 < K; k0 += 32) {
        __builtin_amdgcn_s_wait_tensorcnt(0);    // own subtile of current k0 done
        __syncthreads();                         // everyone's subtiles done
        if (k0 + 32 < K)                         // overlap: DMA next tile
            tdm_load_tile32x32(gbase + k0 + 32, lds_base[buf ^ 1], K);

        v16bf af[2], bfr[4];
        #pragma unroll
        for (int mi = 0; mi < 2; ++mi) {
            const bf16* p = &sA[buf][wm * 32 + mi * 16 + l16][half * 8];
            af[mi] = make_frag(p, p + 16);
        }
        #pragma unroll
        for (int ni = 0; ni < 4; ++ni) {
            const bf16* p = &sB[buf][wn * 64 + ni * 16 + l16][half * 8];
            bfr[ni] = make_frag(p, p + 16);
        }
        #pragma unroll
        for (int mi = 0; mi < 2; ++mi)
            #pragma unroll
            for (int ni = 0; ni < 4; ++ni)
                acc[mi][ni] = wmma_bf16(af[mi], bfr[ni], acc[mi][ni]);
        buf ^= 1;
    }
#else
    __shared__ bf16 sA[128][40];
    __shared__ bf16 sB[128][40];
    const int lrow = tid >> 1;
    const int lseg = (tid & 1) * 16;

    for (int k0 = 0; k0 < K; k0 += 32) {
        const bf16* ga = A  + (size_t)(row0 + lrow) * K + k0 + lseg;
        const bf16* gb = BT + (size_t)(col0 + lrow) * K + k0 + lseg;
        ((uint4*)&sA[lrow][lseg])[0] = ((const uint4*)ga)[0];
        ((uint4*)&sA[lrow][lseg + 8])[0] = ((const uint4*)ga)[1];
        ((uint4*)&sB[lrow][lseg])[0] = ((const uint4*)gb)[0];
        ((uint4*)&sB[lrow][lseg + 8])[0] = ((const uint4*)gb)[1];
        if (k0 + 32 < K) {
            __builtin_prefetch(ga + 32, 0, 1);
            __builtin_prefetch(gb + 32, 0, 1);
        }
        __syncthreads();

        v16bf af[2], bfr[4];
        #pragma unroll
        for (int mi = 0; mi < 2; ++mi) {
            const bf16* p = &sA[wm * 32 + mi * 16 + l16][half * 8];
            af[mi] = make_frag(p, p + 16);
        }
        #pragma unroll
        for (int ni = 0; ni < 4; ++ni) {
            const bf16* p = &sB[wn * 64 + ni * 16 + l16][half * 8];
            bfr[ni] = make_frag(p, p + 16);
        }
        #pragma unroll
        for (int mi = 0; mi < 2; ++mi)
            #pragma unroll
            for (int ni = 0; ni < 4; ++ni)
                acc[mi][ni] = wmma_bf16(af[mi], bfr[ni], acc[mi][ni]);
        __syncthreads();
    }
#endif

    #pragma unroll
    for (int mi = 0; mi < 2; ++mi)
        #pragma unroll
        for (int ni = 0; ni < 4; ++ni) {
            const int c = col0 + wn * 64 + ni * 16 + l16;
            const float bv = bias[c];
            #pragma unroll
            for (int i = 0; i < 8; ++i) {
                const int r = row0 + wm * 32 + mi * 16 + i + 8 * half;
                C[(size_t)r * N + c] = acc[mi][ni][i] + bv;
            }
        }
}

// ---------------------------------------------------------------------------
// 4) RoPE on q,k + repack qkv -> [head][t][hd] bf16
// ---------------------------------------------------------------------------
__global__ __launch_bounds__(256) void rope_kernel(
    const float* __restrict__ tbuf, const float* __restrict__ sin_,
    const float* __restrict__ cos_, bf16* __restrict__ q,
    bf16* __restrict__ k, bf16* __restrict__ v)
{
    const int idx = blockIdx.x * 256 + threadIdx.x;   // (t, h, pair)
    const int i  = idx & 63;
    const int h  = (idx >> 6) & (NH - 1);
    const int tt = idx >> 10;
    if (tt >= T_DIM) return;
    const int d0 = 2 * i, d1 = 2 * i + 1;
    const float c0 = cos_[tt * HD + d0], c1 = cos_[tt * HD + d1];
    const float s0 = sin_[tt * HD + d0], s1 = sin_[tt * HD + d1];
    const float* tr = tbuf + (size_t)tt * NT_COLS + 2 * I_DIM;
    const float q0 = tr[(0 * NH + h) * HD + d0], q1 = tr[(0 * NH + h) * HD + d1];
    const float k0 = tr[(1 * NH + h) * HD + d0], k1 = tr[(1 * NH + h) * HD + d1];
    const float v0 = tr[(2 * NH + h) * HD + d0], v1 = tr[(2 * NH + h) * HD + d1];
    const size_t o = ((size_t)h * T_DIM + tt) * HD;
    q[o + d0] = (bf16)(q0 * c0 - q1 * s0);
    q[o + d1] = (bf16)(q1 * c1 + q0 * s1);
    k[o + d0] = (bf16)(k0 * c0 - k1 * s0);
    k[o + d1] = (bf16)(k1 * c1 + k0 * s1);
    v[o + d0] = (bf16)v0;
    v[o + d1] = (bf16)v1;
}

// ---------------------------------------------------------------------------
// 5) Flash attention, WMMA bf16.  grid=(T/64, NH), block=128 (4 waves).
// ---------------------------------------------------------------------------
__global__ __launch_bounds__(128) void attn_kernel(
    const bf16* __restrict__ qb, const bf16* __restrict__ kb,
    const bf16* __restrict__ vb, const float* __restrict__ bias,
    bf16* __restrict__ comb)
{
    __shared__ bf16 ksh[32][136];     // K tile row-major [s][hd]
    __shared__ bf16 vsh[128][40];     // V tile transposed [hd][s]
    __shared__ bf16 psh[4][16][40];   // per-wave P staging

    const int tid = threadIdx.x, lane = tid & 31, wid = tid >> 5;
    const int half = lane >> 4, l16 = lane & 15;
    const int h = blockIdx.y;
    const int rowW = blockIdx.x * 64 + wid * 16;

    v16bf qf[4];
    {
        const bf16* qrow = qb + ((size_t)h * T_DIM + rowW + l16) * HD;
        #pragma unroll
        for (int kc = 0; kc < 4; ++kc)
            qf[kc] = make_frag(qrow + kc * 32 + half * 8,
                               qrow + kc * 32 + 16 + half * 8);
    }

    v8f o[8];
    #pragma unroll
    for (int ht = 0; ht < 8; ++ht) o[ht] = (v8f){0,0,0,0,0,0,0,0};
    float mrun[8], lrun[8];
    #pragma unroll
    for (int i = 0; i < 8; ++i) { mrun[i] = -3.0e38f; lrun[i] = 0.f; }

    const float scale = 0.08838834764831845f;   // 1/sqrt(128)

    for (int s0 = 0; s0 < T_DIM; s0 += 32) {
        {
            const int r = tid >> 2, sg = (tid & 3) * 32;
            const bf16* gk = kb + ((size_t)h * T_DIM + s0 + r) * HD + sg;
            #pragma unroll
            for (int b = 0; b < 4; ++b)
                ((uint4*)&ksh[r][sg])[b] = ((const uint4*)gk)[b];
            const bf16* gv = vb + ((size_t)h * T_DIM + s0 + r) * HD + sg;
            #pragma unroll
            for (int b = 0; b < 4; ++b) {
                v8bf vv = *(const v8bf*)(gv + b * 8);
                #pragma unroll
                for (int j = 0; j < 8; ++j) vsh[sg + b * 8 + j][r] = vv[j];
            }
        }
        __syncthreads();

        v8f s[2];
        s[0] = (v8f){0,0,0,0,0,0,0,0};
        s[1] = (v8f){0,0,0,0,0,0,0,0};
        #pragma unroll
        for (int nt = 0; nt < 2; ++nt)
            #pragma unroll
            for (int kc = 0; kc < 4; ++kc) {
                const bf16* p = &ksh[nt * 16 + l16][kc * 32 + half * 8];
                s[nt] = wmma_bf16(qf[kc], make_frag(p, p + 16), s[nt]);
            }

        #pragma unroll
        for (int i = 0; i < 8; ++i) {
            const int rr = rowW + i + 8 * half;
            const size_t brow = ((size_t)h * T_DIM + rr) * T_DIM + s0;
            s[0][i] = s[0][i] * scale + bias[brow + l16];
            s[1][i] = s[1][i] * scale + bias[brow + 16 + l16];
            float mx = fmaxf(s[0][i], s[1][i]);
            #pragma unroll
            for (int d = 8; d >= 1; d >>= 1)
                mx = fmaxf(mx, __shfl_xor(mx, d, 32));
            const float mnew = fmaxf(mrun[i], mx);
            const float corr = __expf(mrun[i] - mnew);
            mrun[i] = mnew;
            lrun[i] *= corr;
            #pragma unroll
            for (int ht = 0; ht < 8; ++ht) o[ht][i] *= corr;
            const float p0 = __expf(s[0][i] - mnew);
            const float p1 = __expf(s[1][i] - mnew);
            float rs = p0 + p1;
            #pragma unroll
            for (int d = 8; d >= 1; d >>= 1)
                rs += __shfl_xor(rs, d, 32);
            lrun[i] += rs;
            psh[wid][i + 8 * half][l16]      = (bf16)p0;
            psh[wid][i + 8 * half][16 + l16] = (bf16)p1;
        }

        {
            const bf16* pp = &psh[wid][l16][half * 8];
            const v16bf pa = make_frag(pp, pp + 16);
            #pragma unroll
            for (int ht = 0; ht < 8; ++ht) {
                const bf16* vp = &vsh[ht * 16 + l16][half * 8];
                o[ht] = wmma_bf16(pa, make_frag(vp, vp + 16), o[ht]);
            }
        }
        __syncthreads();
    }

    #pragma unroll
    for (int ht = 0; ht < 8; ++ht) {
        const int c = h * HD + ht * 16 + l16;
        #pragma unroll
        for (int i = 0; i < 8; ++i) {
            const int r = rowW + i + 8 * half;
            comb[(size_t)r * COMB_COLS + c] = (bf16)(o[ht][i] / lrun[i]);
        }
    }
}

// ---------------------------------------------------------------------------
// 6) SiLU gate -> ff part of `combined`
// ---------------------------------------------------------------------------
__global__ __launch_bounds__(256) void silu_kernel(
    const float* __restrict__ tbuf, bf16* __restrict__ comb)
{
    const int idx = blockIdx.x * 256 + threadIdx.x;
    const int row = idx >> 13;
    const int i   = idx & (I_DIM - 1);
    if (row >= T_DIM) return;
    const float a = tbuf[(size_t)row * NT_COLS + i];
    const float b = tbuf[(size_t)row * NT_COLS + I_DIM + i];
    const float s = a / (1.0f + __expf(-a));
    comb[(size_t)row * COMB_COLS + H_DIM + i] = (bf16)(s * b);
}

// ---------------------------------------------------------------------------
// launch
// ---------------------------------------------------------------------------
extern "C" void kernel_launch(void* const* d_in, const int* in_sizes, int n_in,
                              void* d_out, int out_size, void* d_ws, size_t ws_size,
                              hipStream_t stream)
{
    (void)in_sizes; (void)n_in; (void)out_size; (void)ws_size;
    const float* x      = (const float*)d_in[0];
    const float* ages   = (const float*)d_in[1];
    const float* sin_   = (const float*)d_in[2];
    const float* cos_   = (const float*)d_in[3];
    const float* bias   = (const float*)d_in[4];
    const float* norm_w = (const float*)d_in[5];
    const float* w_in   = (const float*)d_in[6];
    const float* b_in   = (const float*)d_in[7];
    const float* w_out  = (const float*)d_in[8];
    const float* b_out  = (const float*)d_in[9];
    float* out = (float*)d_out;

    char* ws = (char*)d_ws;
    bf16*  xn    = (bf16*)(ws + XN_OFF);
    bf16*  winT  = (bf16*)(ws + WINT_OFF);
    bf16*  woutT = (bf16*)(ws + WOUTT_OFF);
    float* tbuf  = (float*)(ws + TBUF_OFF);
    bf16*  qw    = (bf16*)(ws + QKV_OFF);
    bf16*  kw    = (bf16*)(ws + QKV_OFF + QKV_ONE);
    bf16*  vw    = (bf16*)(ws + QKV_OFF + 2 * QKV_ONE);
    bf16*  comb  = (bf16*)(ws + COMB_OFF);

    rmsnorm_kernel<<<T_DIM, 256, 0, stream>>>(x, ages, norm_w, xn);

    transpose_conv_kernel<<<dim3(NT_COLS / 32, H_DIM / 32), dim3(32, 8), 0, stream>>>(
        w_in, winT, H_DIM, NT_COLS);
    transpose_conv_kernel<<<dim3(H_DIM / 32, COMB_COLS / 32), dim3(32, 8), 0, stream>>>(
        w_out, woutT, COMB_COLS, H_DIM);

    gemm_bf16_kernel<<<dim3(NT_COLS / 128, T_DIM / 128), 256, 0, stream>>>(
        xn, winT, b_in, tbuf, T_DIM, NT_COLS, H_DIM);

    rope_kernel<<<(T_DIM * NH * 64) / 256, 256, 0, stream>>>(
        tbuf, sin_, cos_, qw, kw, vw);

    attn_kernel<<<dim3(T_DIM / 64, NH), 128, 0, stream>>>(qw, kw, vw, bias, comb);

    silu_kernel<<<(T_DIM * I_DIM) / 256, 256, 0, stream>>>(tbuf, comb);

    gemm_bf16_kernel<<<dim3(H_DIM / 128, T_DIM / 128), 256, 0, stream>>>(
        comb, woutT, b_out, out, T_DIM, H_DIM, COMB_COLS);
}
